// GQA_66056597013126
// MI455X (gfx1250) — matrix operand order repaired and checked by reading, not compiled, not run on hardware
//
#include <hip/hip_runtime.h>
#include <hip/hip_bf16.h>

typedef __attribute__((ext_vector_type(16))) _Float16 v16h;
typedef __attribute__((ext_vector_type(8)))  float    v8f;
typedef __attribute__((ext_vector_type(4)))  unsigned int u32x4;
typedef __attribute__((ext_vector_type(8)))  int      i32x8;
typedef __attribute__((ext_vector_type(4)))  int      i32x4;

enum StoreMode { ST_F16_ROW = 0, ST_F16_VT = 1, ST_F32_ROW = 2 };

namespace {

constexpr int D_MODEL   = 1024;
constexpr int NUM_HEADS = 16;
constexpr int NUM_GROUPS= 4;
constexpr int D_K       = 64;
constexpr int KV_DIM    = 256;   // NUM_GROUPS * D_K
constexpr int B_        = 2;
constexpr int S_        = 2048;
constexpr int MTOK      = B_ * S_;   // 4096 tokens
constexpr float SCALE   = 0.125f;    // 1/sqrt(64)

__device__ __forceinline__ v8f wmma_f16(v16h a, v16h b, v8f c) {
  // D = A(16x32 f16) * B(32x16 f16) + C(16x16 f32)
  return __builtin_amdgcn_wmma_f32_16x16x32_f16(
      /*neg_a=*/false, a, /*neg_b=*/false, b,
      /*c_mod=*/(short)0, c, /*reuse_a=*/false, /*reuse_b=*/false);
}

// K-index pattern for 16-bit A-matrix 16x32 fragment (contiguous, matches B):
// lane 0-15 row m=lane holds K {0..15}; lane 16-31 holds K {16..31}
__device__ __forceinline__ int a_kidx(int e, int lane) {
  return (((lane >> 4) & 1) << 4) + e;
}

// ---------------------------------------------------------------------------
// Tensor Data Mover: DMA a 2D fp32 tile (tile_w x tile_h elements) from global
// into LDS, inserting 2 pad DWORDs after every 64 DWORDs (row stride 66 f32).
// D# layout per CDNA5 ISA.
// ---------------------------------------------------------------------------
__device__ __forceinline__ void tdm_load_tile_f32_pad66(
    unsigned int lds_addr, const void* gptr,
    unsigned int tensor_w, unsigned int tensor_h,
    unsigned int tile_w, unsigned int tile_h, unsigned int row_stride_elems)
{
  const unsigned long long ga = (unsigned long long)(uintptr_t)gptr;

  u32x4 g0;
  g0[0] = 1u;                                   // count=1 (valid user D#)
  g0[1] = lds_addr;                             // lds_addr [63:32]
  g0[2] = (unsigned int)ga;                     // global_addr [95:64]
  g0[3] = (unsigned int)((ga >> 32) & 0x1FFFFFFu) | (2u << 30);  // addr hi | type=2

  i32x8 g1;
  // w0: workgroup_mask=0 | data_size=2 (4B) | pad_enable | pad_interval=5 (64 dw)
  //     | pad_amount=1 (2 dwords)
  g1[0] = (int)((2u << 16) | (1u << 20) | (5u << 22) | (1u << 25));
  g1[1] = (int)((tensor_w & 0xFFFFu) << 16);                       // tensor_dim0 lo
  g1[2] = (int)((tensor_w >> 16) | ((tensor_h & 0xFFFFu) << 16));  // dim0 hi | dim1 lo
  g1[3] = (int)((tensor_h >> 16) | (tile_w << 16));                // dim1 hi | tile_dim0
  g1[4] = (int)tile_h;                                             // tile_dim1 (tile_dim2=0)
  g1[5] = (int)row_stride_elems;                                   // tensor_dim0_stride lo
  g1[6] = 0;                                                       // stride0 hi | stride1 lo
  g1[7] = 0;                                                       // stride1 hi

  i32x4 z4 = {0, 0, 0, 0};
#if defined(__clang_major__) && (__clang_major__ >= 23)
  i32x8 z8 = {0, 0, 0, 0, 0, 0, 0, 0};
  __builtin_amdgcn_tensor_load_to_lds(g0, g1, z4, z4, z8, 0);
#else
  __builtin_amdgcn_tensor_load_to_lds(g0, g1, z4, z4, 0);
#endif
}

} // namespace

// ---------------------------------------------------------------------------
// Generic tiled GEMM: C[M,N] = A[M,K] * W[K,N] + bias, via WMMA f16->f32.
// 256 threads = 8 waves in 4(M) x 2(N); WG tile 64x64.
// A staged to LDS manually (fp32->f16 convert). W tile DMA'd by the TDM into
// LDS as raw fp32 with padded rows (66 f32 -> conflict-free column reads);
// converted to f16 during B-fragment assembly.
// ---------------------------------------------------------------------------
template <bool A_HALF>
__global__ __launch_bounds__(256)
void gemm_wmma_kernel(const void* __restrict__ Ap, const float* __restrict__ W,
                      const float* __restrict__ bias, void* __restrict__ Cp,
                      int M, int N, int K, int storeMode)
{
  __shared__ __align__(32) _Float16 Asf[64][36];  // [m][k], padded rows
  __shared__ __align__(16) float    Wt32[32][66]; // [k][n], TDM-padded rows

  const int tid  = threadIdx.x;
  const int lane = tid & 31;
  const int w    = tid >> 5;
  const int wm   = w >> 1;        // 0..3
  const int wn   = w & 1;         // 0..1
  const int m0   = blockIdx.y * 64;
  const int n0   = blockIdx.x * 64;

  const unsigned int ldsW = (unsigned int)(uintptr_t)&Wt32[0][0]; // low 32b = LDS addr

  v8f acc0 = {}; v8f acc1 = {};

  for (int k0 = 0; k0 < K; k0 += 32) {
    // ---- TDM: DMA W slab (32 rows x 64 cols fp32) into LDS (wave 0 issues)
    if (w == 0) {
      tdm_load_tile_f32_pad66(ldsW, W + (size_t)k0 * N + n0,
                              (unsigned)(N - n0), (unsigned)(K - k0),
                              64u, 32u, (unsigned)N);
    }
    // ---- stage A tile (64 x 32) -> f16 LDS (overlaps with the DMA)
    {
      const int i = tid * 8;
      const int r = i >> 5, c = i & 31;
      if (A_HALF) {
        const _Float16* src = (const _Float16*)Ap + (size_t)(m0 + r) * K + k0 + c;
#pragma unroll
        for (int j = 0; j < 8; ++j) Asf[r][c + j] = src[j];
      } else {
        const float* src = (const float*)Ap + (size_t)(m0 + r) * K + k0 + c;
#pragma unroll
        for (int j = 0; j < 8; ++j) Asf[r][c + j] = (_Float16)src[j];
      }
    }
    // prefetch next A K-slab (global_prefetch_b8)
    if (k0 + 32 < K) {
      __builtin_prefetch((const char*)Ap + ((size_t)m0 * K + k0 + 32) *
                         (A_HALF ? sizeof(_Float16) : sizeof(float)), 0, 1);
    }
    if (w == 0) __builtin_amdgcn_s_wait_tensorcnt(0);  // issuing wave drains TDM
    __syncthreads();

    // ---- A fragment (16x32) for this wave's M sub-tile
    v16h a;
    {
      const int mr = wm * 16 + (lane & 15);
#pragma unroll
      for (int e = 0; e < 16; ++e) a[e] = Asf[mr][a_kidx(e, lane)];
    }
    // ---- two B fragments (32x16): column reads from TDM tile, cvt to f16
    const int khB = ((lane >> 4) & 1) << 4;   // lanes 0-15: K 0-15, lanes 16-31: 16-31
    const int nc  = wn * 32 + (lane & 15);
    v16h b0, b1;
#pragma unroll
    for (int e = 0; e < 16; ++e) {
      b0[e] = (_Float16)Wt32[khB + e][nc];
      b1[e] = (_Float16)Wt32[khB + e][nc + 16];
    }
    acc0 = wmma_f16(a, b0, acc0);
    acc1 = wmma_f16(a, b1, acc1);
    __syncthreads();
  }

  // ---- epilogue: C/D layout: vgpr v -> M = v + 8*(lane>=16), N = lane&15
  const int mbase = m0 + wm * 16 + (((lane >> 4) & 1) << 3);
  const int nA = n0 + wn * 32 + (lane & 15);
  const int nB = nA + 16;
  const float biasA = bias[nA];
  const float biasB = bias[nB];

  if (storeMode == ST_F16_ROW) {
    _Float16* C = (_Float16*)Cp;
#pragma unroll
    for (int v = 0; v < 8; ++v) {
      const int m = mbase + v;
      C[(size_t)m * N + nA] = (_Float16)(acc0[v] + biasA);
      C[(size_t)m * N + nB] = (_Float16)(acc1[v] + biasB);
    }
  } else if (storeMode == ST_F16_VT) {
    // V projection stored transposed: Vt[b][g][d][s]
    _Float16* C = (_Float16*)Cp;
#pragma unroll
    for (int v = 0; v < 8; ++v) {
      const int m = mbase + v;                 // token
      const int b = m >> 11, s = m & (S_ - 1);
      const int gA = nA >> 6, dA = nA & 63;
      const int gB = nB >> 6, dB = nB & 63;
      C[((size_t)(b * NUM_GROUPS + gA) * D_K + dA) * S_ + s] = (_Float16)(acc0[v] + biasA);
      C[((size_t)(b * NUM_GROUPS + gB) * D_K + dB) * S_ + s] = (_Float16)(acc1[v] + biasB);
    }
  } else { // ST_F32_ROW
    float* C = (float*)Cp;
#pragma unroll
    for (int v = 0; v < 8; ++v) {
      const int m = mbase + v;
      C[(size_t)m * N + nA] = acc0[v] + biasA;
      C[(size_t)m * N + nB] = acc1[v] + biasB;
    }
  }
}

// ---------------------------------------------------------------------------
// scores[b,h,q,t] = scale * sum_d Q[b,q,h,d] * K[b,t,g,d]
// WG = 64q x 64t per (b,h); 8 waves (4q x 2t), dk=64 -> 2 K-steps of 32.
// Fragments loaded straight from global f16 (compiler merges to b128s).
// ---------------------------------------------------------------------------
__global__ __launch_bounds__(256)
void scores_kernel(const _Float16* __restrict__ Qf, const _Float16* __restrict__ Kf,
                   float* __restrict__ attn)
{
  const int tid = threadIdx.x, lane = tid & 31, w = tid >> 5;
  const int wm = w >> 1, wn = w & 1;
  const int bh = blockIdx.z;
  const int b = bh >> 4, h = bh & 15, g = h >> 2;
  const int q0 = blockIdx.y * 64 + wm * 16;
  const int t0 = blockIdx.x * 64 + wn * 32;

  const int qrow = b * S_ + q0 + (lane & 15);
  const int trow = b * S_ + t0 + (lane & 15);
  const int khB  = ((lane >> 4) & 1) << 4;

  v8f acc0 = {}; v8f acc1 = {};
#pragma unroll
  for (int ks = 0; ks < 2; ++ks) {
    const _Float16* qp = Qf + (size_t)qrow * D_MODEL + h * D_K + ks * 32;
    v16h a;
#pragma unroll
    for (int e = 0; e < 16; ++e) a[e] = qp[a_kidx(e, lane)];

    // B column n = key token; contiguous 16 halves over d -> one 32B load
    const _Float16* kp = Kf + (size_t)trow * KV_DIM + g * D_K + ks * 32 + khB;
    const v16h b0 = *(const v16h*)kp;
    acc0 = wmma_f16(a, b0, acc0);
    const v16h b1 = *(const v16h*)(kp + (size_t)16 * KV_DIM);
    acc1 = wmma_f16(a, b1, acc1);
  }

  const int mbase = q0 + (((lane >> 4) & 1) << 3);
  float* outp = attn + (size_t)bh * S_ * S_;
#pragma unroll
  for (int v = 0; v < 8; ++v) {
    const size_t row = (size_t)(mbase + v) * S_;
    outp[row + t0 + (lane & 15)]      = acc0[v] * SCALE;
    outp[row + t0 + 16 + (lane & 15)] = acc1[v] * SCALE;
  }
}

// ---------------------------------------------------------------------------
// In-place softmax over each length-2048 row of attn. One 256-thread WG/row,
// 8 elems per thread via float4, LDS tree reductions for max and sum.
// ---------------------------------------------------------------------------
__global__ __launch_bounds__(256)
void softmax_kernel(float* __restrict__ attn)
{
  __shared__ float red[256];
  float4* p4 = (float4*)(attn + (size_t)blockIdx.x * S_);
  const int t = threadIdx.x;

  float4 x0 = p4[t];
  float4 x1 = p4[t + 256];

  float m = fmaxf(fmaxf(fmaxf(x0.x, x0.y), fmaxf(x0.z, x0.w)),
                  fmaxf(fmaxf(x1.x, x1.y), fmaxf(x1.z, x1.w)));
  red[t] = m; __syncthreads();
#pragma unroll
  for (int s = 128; s > 0; s >>= 1) {
    if (t < s) red[t] = fmaxf(red[t], red[t + s]);
    __syncthreads();
  }
  m = red[0]; __syncthreads();

  x0.x = __expf(x0.x - m); x0.y = __expf(x0.y - m);
  x0.z = __expf(x0.z - m); x0.w = __expf(x0.w - m);
  x1.x = __expf(x1.x - m); x1.y = __expf(x1.y - m);
  x1.z = __expf(x1.z - m); x1.w = __expf(x1.w - m);

  red[t] = (x0.x + x0.y + x0.z + x0.w) + (x1.x + x1.y + x1.z + x1.w);
  __syncthreads();
#pragma unroll
  for (int s = 128; s > 0; s >>= 1) {
    if (t < s) red[t] += red[t + s];
    __syncthreads();
  }
  const float inv = 1.0f / red[0];

  x0.x *= inv; x0.y *= inv; x0.z *= inv; x0.w *= inv;
  x1.x *= inv; x1.y *= inv; x1.z *= inv; x1.w *= inv;
  p4[t] = x0; p4[t + 256] = x1;
}

// ---------------------------------------------------------------------------
// O[b,q,h,d] = sum_t attn[b,h,q,t] * V[b,t,g,d]  using Vt[b,g,d,t].
// WG = 16q x 64d per (b,h); 8 waves: split-K=2 halves x 4 d-tiles of 16;
// 32 wmma K-steps per half; halves reduced through LDS.
// ---------------------------------------------------------------------------
__global__ __launch_bounds__(256)
void pv_kernel(const float* __restrict__ attn, const _Float16* __restrict__ Vt,
               _Float16* __restrict__ O)
{
  __shared__ float red[16][64];
  const int tid = threadIdx.x, lane = tid & 31, w = tid >> 5;
  const int half = w >> 2, nt = w & 3;
  const int q0 = blockIdx.x * 16;
  const int h = blockIdx.y, b = blockIdx.z, g = h >> 2;

  const float* arow = attn +
      ((size_t)(b * NUM_HEADS + h) * S_ + q0 + (lane & 15)) * S_;
  const _Float16* vrow = Vt +
      ((size_t)(b * NUM_GROUPS + g) * D_K + nt * 16 + (lane & 15)) * S_ +
      (((lane >> 4) & 1) << 4);

  v8f acc = {};
  const int ks0 = half * 32;
#pragma unroll 4
  for (int ks = ks0; ks < ks0 + 32; ++ks) {
    const float* ap = arow + ks * 32;
    v16h a;
#pragma unroll
    for (int e = 0; e < 16; ++e) a[e] = (_Float16)ap[a_kidx(e, lane)];
    const v16h bb = *(const v16h*)(vrow + ks * 32);   // contiguous 32B over t
    acc = wmma_f16(a, bb, acc);
  }

  const int mb = ((lane >> 4) & 1) << 3;
  const int d  = nt * 16 + (lane & 15);
  if (half == 1) {
#pragma unroll
    for (int v = 0; v < 8; ++v) red[mb + v][d] = acc[v];
  }
  __syncthreads();
  if (half == 0) {
#pragma unroll
    for (int v = 0; v < 8; ++v) {
      const int m = mb + v;
      const float val = acc[v] + red[m][d];
      O[(size_t)(b * S_ + q0 + m) * D_MODEL + h * D_K + d] = (_Float16)val;
    }
  }
}

// ---------------------------------------------------------------------------
extern "C" void kernel_launch(void* const* d_in, const int* in_sizes, int n_in,
                              void* d_out, int out_size, void* d_ws, size_t ws_size,
                              hipStream_t stream)
{
  (void)in_sizes; (void)n_in; (void)out_size; (void)ws_size;

  const float* query = (const float*)d_in[0];
  const float* key_  = (const float*)d_in[1];
  const float* value = (const float*)d_in[2];
  const float* Wq = (const float*)d_in[3];
  const float* bq = (const float*)d_in[4];
  const float* Wk = (const float*)d_in[5];
  const float* bk = (const float*)d_in[6];
  const float* Wv = (const float*)d_in[7];
  const float* bv = (const float*)d_in[8];
  const float* Wo = (const float*)d_in[9];
  const float* bo = (const float*)d_in[10];

  float* out  = (float*)d_out;                          // [B,S,1024] fp32
  float* attn = out + (size_t)MTOK * D_MODEL;           // [B,H,S,S]  fp32

  char* ws = (char*)d_ws;
  _Float16* Qf = (_Float16*)ws;                          // [4096,1024] f16 (8 MiB)
  _Float16* Kf = (_Float16*)(ws + (8ull  << 20));        // [4096,256]  f16 (2 MiB)
  _Float16* Vt = (_Float16*)(ws + (10ull << 20));        // [B,G,64,S]  f16 (2 MiB)
  _Float16* Of = (_Float16*)(ws + (12ull << 20));        // [4096,1024] f16 (8 MiB)

  const dim3 blk(256);

  // Q / K / V projections (V stored transposed for PV B-fragments)
  gemm_wmma_kernel<false><<<dim3(D_MODEL / 64, MTOK / 64), blk, 0, stream>>>(
      query, Wq, bq, Qf, MTOK, D_MODEL, D_MODEL, ST_F16_ROW);
  gemm_wmma_kernel<false><<<dim3(KV_DIM / 64, MTOK / 64), blk, 0, stream>>>(
      key_, Wk, bk, Kf, MTOK, KV_DIM, D_MODEL, ST_F16_ROW);
  gemm_wmma_kernel<false><<<dim3(KV_DIM / 64, MTOK / 64), blk, 0, stream>>>(
      value, Wv, bv, Vt, MTOK, KV_DIM, D_MODEL, ST_F16_VT);

  // scores = Q K^T * scale  -> attn region (raw)
  scores_kernel<<<dim3(S_ / 64, S_ / 64, B_ * NUM_HEADS), blk, 0, stream>>>(
      Qf, Kf, attn);

  // row softmax in place
  softmax_kernel<<<dim3(B_ * NUM_HEADS * S_), blk, 0, stream>>>(attn);

  // O = attn @ V (head-major f16)
  pv_kernel<<<dim3(S_ / 16, NUM_HEADS, B_), blk, 0, stream>>>(attn, Vt, Of);

  // final projection: out = O @ Wo + bo (fp32)
  gemm_wmma_kernel<true><<<dim3(D_MODEL / 64, MTOK / 64), blk, 0, stream>>>(
      Of, Wo, bo, out, MTOK, D_MODEL, D_MODEL, ST_F32_ROW);
}
